// quantized_conv_38938173506080
// MI455X (gfx1250) — 8-bit matrix operand order repaired and checked by reading, compiled, not run on hardware
//
#include <hip/hip_runtime.h>


typedef __attribute__((ext_vector_type(8))) int v8i;

#define F_TOT 576
#define M_TOT 16384
#define O_TOT 64
#define OUT_ELEMS 1048576
#define APLANE 9437184u
#define BPLANE 36864u


__device__ __forceinline__ v8i loadA_frag(const unsigned char* p) {
  const uint2* q = (const uint2*)p;
  const uint2 a0 = q[0], a1 = q[1], a2 = q[4], a3 = q[5];
  v8i r;
  r[0] = (int)a0.x; r[1] = (int)a0.y; r[2] = (int)a1.x; r[3] = (int)a1.y;
  r[4] = (int)a2.x; r[5] = (int)a2.y; r[6] = (int)a3.x; r[7] = (int)a3.y;
  return r;
}

__device__ __forceinline__ v8i loadB_frag(const unsigned char* p) {
  const uint4* q = (const uint4*)p;
  const uint4 b0 = q[0], b1 = q[2];
  v8i r;
  r[0] = (int)b0.x; r[1] = (int)b0.y; r[2] = (int)b0.z; r[3] = (int)b0.w;
  r[4] = (int)b1.x; r[5] = (int)b1.y; r[6] = (int)b1.z; r[7] = (int)b1.w;
  return r;
}

__device__ __forceinline__ v8i unpack_frag(const unsigned int* raw, int shift) {
  v8i r;
#pragma unroll
  for (int i = 0; i < 8; ++i)
    r[i] = (int)((raw[i] >> shift) & 0x03030303u);
  return r;
}

__device__ __forceinline__ void loadA8(unsigned int* r, const unsigned char* p) {
  const uint2* q = (const uint2*)p;
  const uint2 a0 = q[0], a1 = q[1], a2 = q[4], a3 = q[5];
  r[0] = a0.x; r[1] = a0.y; r[2] = a1.x; r[3] = a1.y;
  r[4] = a2.x; r[5] = a2.y; r[6] = a3.x; r[7] = a3.y;
}

__device__ __forceinline__ void loadB8(unsigned int* r, const unsigned char* p) {
  const uint4* q = (const uint4*)p;
  const uint4 b0 = q[0], b1 = q[2];
  r[0] = b0.x; r[1] = b0.y; r[2] = b0.z; r[3] = b0.w;
  r[4] = b1.x; r[5] = b1.y; r[6] = b1.z; r[7] = b1.w;
}


__device__ __forceinline__ int quant_input(const float* __restrict__ x,
                                           unsigned int idx) {
  const int f = (int)(idx % F_TOT);
  const int m = (int)(idx / F_TOT);
  const int b = m >> 10, p = m & 1023;
  const int h = p >> 5, wcol = p & 31;
  const int c = f / 9, rr = f % 9;
  const int ky = rr / 3, kx = rr % 3;
  const int hi = h + ky - 1, wi = wcol + kx - 1;
  float val = 0.f;
  if (hi >= 0 && hi < 32 && wi >= 0 && wi < 32)
    val = x[(((b * 64 + c) * 32 + hi) * 32) + wi];
  const float cl = fminf(fmaxf(val, -2.0f), 1.984375f);
  return ((int)rintf(cl * 64.f)) & 255;
}


__global__ void __launch_bounds__(256) prep_weights_kernel(
    const float* __restrict__ w, unsigned char* __restrict__ Qp,
    unsigned char* __restrict__ Qn, float* __restrict__ meta,
    int* __restrict__ maxpos, int* __restrict__ maxneg,
    float* __restrict__ out, int expand) {
  __shared__ float red[256];
  const int tid = threadIdx.x;
  float lmax = 0.f;
  for (int i = tid; i < O_TOT * F_TOT; i += 256) lmax = fmaxf(lmax, fabsf(w[i]));
  red[tid] = lmax;
  __syncthreads();
  for (int off = 128; off > 0; off >>= 1) {
    if (tid < off) red[tid] = fmaxf(red[tid], red[tid + off]);
    __syncthreads();
  }
  const float raw_max = red[0];
  const float ma = (raw_max > 0.f) ? raw_max : 1.f;
  if (tid == 0) {
    meta[0] = raw_max;
    maxpos[0] = 0;
    maxneg[0] = 0;
    out[OUT_ELEMS] = 0.f;
  }
  for (int i = tid; i < O_TOT * F_TOT; i += 256) {
    const float wv = w[i];
    const int pi = (int)rintf(fmaxf(wv, 0.f) / ma * 255.f);
    const int ni = (int)rintf(fmaxf(-wv, 0.f) / ma * 255.f);
    if (expand) {
#pragma unroll
      for (int t = 0; t < 4; ++t) {
        Qp[t * BPLANE + i] = (unsigned char)((pi >> (2 * t)) & 3);
        Qn[t * BPLANE + i] = (unsigned char)((ni >> (2 * t)) & 3);
      }
    } else {
      Qp[i] = (unsigned char)pi;
      Qn[i] = (unsigned char)ni;
    }
  }
}

__global__ void __launch_bounds__(256) prep_inputs_packed_kernel(
    const float* __restrict__ x, unsigned char* __restrict__ Aq) {
  const unsigned int idx = blockIdx.x * 256u + threadIdx.x;
  Aq[idx] = (unsigned char)quant_input(x, idx);
}

__global__ void __launch_bounds__(256) prep_inputs_exp_kernel(
    const float* __restrict__ x, unsigned char* __restrict__ Aexp) {
  const unsigned int idx = blockIdx.x * 256u + threadIdx.x;
  const int q = quant_input(x, idx);
#pragma unroll
  for (int s = 0; s < 4; ++s)
    Aexp[s * APLANE + idx] = (unsigned char)((q >> (2 * s)) & 3);
}


struct AdcParams {
  float stepP, stepN, rinvP, rinvN, outscale;
};

__device__ __forceinline__ AdcParams load_adc(const int* maxpos, const int* maxneg,
                                              const float* meta, int mode) {
  AdcParams a = {0.f, 0.f, 0.f, 0.f, 0.f};
  if (mode != 0) {
    a.stepP = fmaxf((float)maxpos[0], 1e-8f) * (1.0f / 15.0f);
    a.stepN = fmaxf((float)maxneg[0], 1e-8f) * (1.0f / 15.0f);
    a.rinvP = 1.0f / a.stepP;
    a.rinvN = 1.0f / a.stepN;
    a.outscale = meta[0] * (1.0f / 255.0f);
  }
  return a;
}


__global__ void __launch_bounds__(256) bitserial_gemm_exp_kernel(
    const unsigned char* __restrict__ Aexp, const unsigned char* __restrict__ Bp,
    const unsigned char* __restrict__ Bn, int* __restrict__ maxpos,
    int* __restrict__ maxneg, const float* __restrict__ meta,
    float* __restrict__ out, int mode) {
  const int tid = threadIdx.x;
  const int lane = tid & 31;
  const int wid = blockIdx.x * 8 + (tid >> 5);
  const int mtile = wid >> 2;
  const int ntile = wid & 3;
  const int half = lane >> 4;
  const int mrow = mtile * 16 + (lane & 15);
  const int ocol = ntile * 16 + (lane & 15);

  const unsigned char* abase = Aexp + (size_t)mrow * F_TOT + half * 16;
  const unsigned char* pbase = Bp + (size_t)ocol * F_TOT + half * 16;
  const unsigned char* nbase = Bn + (size_t)ocol * F_TOT + half * 16;

  const AdcParams adc = load_adc(maxpos, maxneg, meta, mode);

  float outacc[8];
#pragma unroll
  for (int r = 0; r < 8; ++r) outacc[r] = 0.f;
  int lanemaxP = 0, lanemaxN = 0;

  for (int g = 0; g < 2; ++g) {
    const unsigned char* bbase = g ? nbase : pbase;

    v8i acc[4][4];
    const v8i vzero = {};
#pragma unroll
    for (int s = 0; s < 4; ++s)
#pragma unroll
      for (int t = 0; t < 4; ++t) acc[s][t] = vzero;

    for (int ks = 0; ks < 9; ++ks) {
      v8i a[4], b[4];
#pragma unroll
      for (int s = 0; s < 4; ++s) a[s] = loadA_frag(abase + s * APLANE + ks * 64);
#pragma unroll
      for (int t = 0; t < 4; ++t) b[t] = loadB_frag(bbase + t * BPLANE + ks * 64);
#pragma unroll
      for (int s = 0; s < 4; ++s)
#pragma unroll
        for (int t = 0; t < 4; ++t)
          acc[s][t] = __builtin_amdgcn_wmma_i32_16x16x64_iu8(
              false, a[s], false, b[t], acc[s][t], false, false);
    }

    if (mode == 0) {
#pragma unroll
      for (int s = 0; s < 4; ++s)
#pragma unroll
        for (int t = 0; t < 4; ++t)
#pragma unroll
          for (int r = 0; r < 8; ++r) {
            const int v = acc[s][t][r];
            if (g == 0) lanemaxP = max(lanemaxP, v);
            else        lanemaxN = max(lanemaxN, v);
          }
    } else {
      const float step = g ? adc.stepN : adc.stepP;
      const float rinv = g ? adc.rinvN : adc.rinvP;
      const float sgn = g ? -1.f : 1.f;
#pragma unroll
      for (int s = 0; s < 4; ++s)
#pragma unroll
        for (int t = 0; t < 4; ++t) {
          const float sc = sgn * (float)(1 << (2 * (s + t)));
#pragma unroll
          for (int r = 0; r < 8; ++r)
            outacc[r] += rintf((float)acc[s][t][r] * rinv) * step * sc;
        }
    }
  }

  if (mode == 0) {
    atomicMax(maxpos, lanemaxP);
    atomicMax(maxneg, lanemaxN);
  } else {
    const int mbase = mtile * 16 + 8 * half;
#pragma unroll
    for (int r = 0; r < 8; ++r) {
      const int m = mbase + r;
      const int b = m >> 10, p = m & 1023;
      out[((size_t)b * 64 + ocol) * 1024 + p] = outacc[r] * adc.outscale;
    }
  }
}


__global__ void __launch_bounds__(256) bitserial_gemm_packed_kernel(
    const unsigned char* __restrict__ Aq, const unsigned char* __restrict__ Qp,
    const unsigned char* __restrict__ Qn, int* __restrict__ maxpos,
    int* __restrict__ maxneg, const float* __restrict__ meta,
    float* __restrict__ out, int mode) {
  const int tid = threadIdx.x;
  const int lane = tid & 31;
  const int wid = blockIdx.x * 8 + (tid >> 5);
  const int mtile = wid >> 2;
  const int ntile = wid & 3;
  const int half = lane >> 4;
  const int mrow = mtile * 16 + (lane & 15);
  const int ocol = ntile * 16 + (lane & 15);

  const unsigned char* arow = Aq + (size_t)mrow * F_TOT + half * 16;
  const unsigned char* prow = Qp + (size_t)ocol * F_TOT + half * 16;
  const unsigned char* nrow = Qn + (size_t)ocol * F_TOT + half * 16;

  const AdcParams adc = load_adc(maxpos, maxneg, meta, mode);

  float outacc[8];
#pragma unroll
  for (int r = 0; r < 8; ++r) outacc[r] = 0.f;
  int lanemaxP = 0, lanemaxN = 0;

  for (int g = 0; g < 2; ++g) {
    const unsigned char* brow = g ? nrow : prow;

    v8i acc[4][4];
    const v8i vzero = {};
#pragma unroll
    for (int s = 0; s < 4; ++s)
#pragma unroll
      for (int t = 0; t < 4; ++t) acc[s][t] = vzero;

    unsigned int rawA[8], rawB[8];
    loadA8(rawA, arow);
    loadB8(rawB, brow);

    for (int ks = 0; ks < 9; ++ks) {
      v8i a[4], b[4];
#pragma unroll
      for (int s = 0; s < 4; ++s) a[s] = unpack_frag(rawA, 2 * s);
#pragma unroll
      for (int t = 0; t < 4; ++t) b[t] = unpack_frag(rawB, 2 * t);
      if (ks < 8) {
        loadA8(rawA, arow + (ks + 1) * 64);
        loadB8(rawB, brow + (ks + 1) * 64);
      }
#pragma unroll
      for (int s = 0; s < 4; ++s)
#pragma unroll
        for (int t = 0; t < 4; ++t)
          acc[s][t] = __builtin_amdgcn_wmma_i32_16x16x64_iu8(
              false, a[s], false, b[t], acc[s][t], false, false);
    }

    if (mode == 0) {
#pragma unroll
      for (int s = 0; s < 4; ++s)
#pragma unroll
        for (int t = 0; t < 4; ++t)
#pragma unroll
          for (int r = 0; r < 8; ++r) {
            const int v = acc[s][t][r];
            if (g == 0) lanemaxP = max(lanemaxP, v);
            else        lanemaxN = max(lanemaxN, v);
          }
    } else {
      const float step = g ? adc.stepN : adc.stepP;
      const float rinv = g ? adc.rinvN : adc.rinvP;
      const float sgn = g ? -1.f : 1.f;
#pragma unroll
      for (int s = 0; s < 4; ++s)
#pragma unroll
        for (int t = 0; t < 4; ++t) {
          const float sc = sgn * (float)(1 << (2 * (s + t)));
#pragma unroll
          for (int r = 0; r < 8; ++r)
            outacc[r] += rintf((float)acc[s][t][r] * rinv) * step * sc;
        }
    }
  }

  if (mode == 0) {
    atomicMax(maxpos, lanemaxP);
    atomicMax(maxneg, lanemaxN);
  } else {
    const int mbase = mtile * 16 + 8 * half;
#pragma unroll
    for (int r = 0; r < 8; ++r) {
      const int m = mbase + r;
      const int b = m >> 10, p = m & 1023;
      out[((size_t)b * 64 + ocol) * 1024 + p] = outacc[r] * adc.outscale;
    }
  }
}


extern "C" void kernel_launch(void* const* d_in, const int* in_sizes, int n_in,
                              void* d_out, int out_size, void* d_ws, size_t ws_size,
                              hipStream_t stream) {
  const float* x = (const float*)d_in[0];
  const float* w = (const float*)d_in[1];
  float* out = (float*)d_out;
  unsigned char* ws = (unsigned char*)d_ws;

  const size_t offBp_e = 4ull * APLANE;
  const size_t offBn_e = offBp_e + 4ull * BPLANE;
  const size_t offMeta_e = offBn_e + 4ull * BPLANE;
  const size_t need_exp = offMeta_e + 16;

  if (ws_size >= need_exp) {
    unsigned char* Aexp = ws;
    unsigned char* Bp = ws + offBp_e;
    unsigned char* Bn = ws + offBn_e;
    int* maxpos = (int*)(ws + offMeta_e);
    int* maxneg = (int*)(ws + offMeta_e + 4);
    float* meta = (float*)(ws + offMeta_e + 8);

    prep_weights_kernel<<<1, 256, 0, stream>>>(w, Bp, Bn, meta, maxpos, maxneg, out, 1);
    prep_inputs_exp_kernel<<<(M_TOT * F_TOT) / 256, 256, 0, stream>>>(x, Aexp);
    bitserial_gemm_exp_kernel<<<512, 256, 0, stream>>>(Aexp, Bp, Bn, maxpos, maxneg, meta, out, 0);
    bitserial_gemm_exp_kernel<<<512, 256, 0, stream>>>(Aexp, Bp, Bn, maxpos, maxneg, meta, out, 1);
  } else {
    unsigned char* Aq = ws;
    unsigned char* Qp = ws + APLANE;
    unsigned char* Qn = ws + APLANE + BPLANE;
    int* maxpos = (int*)(ws + APLANE + 2 * BPLANE);
    int* maxneg = (int*)(ws + APLANE + 2 * BPLANE + 4);
    float* meta = (float*)(ws + APLANE + 2 * BPLANE + 8);

    prep_weights_kernel<<<1, 256, 0, stream>>>(w, Qp, Qn, meta, maxpos, maxneg, out, 0);
    prep_inputs_packed_kernel<<<(M_TOT * F_TOT) / 256, 256, 0, stream>>>(x, Aq);
    bitserial_gemm_packed_kernel<<<512, 256, 0, stream>>>(Aq, Qp, Qn, maxpos, maxneg, meta, out, 0);
    bitserial_gemm_packed_kernel<<<512, 256, 0, stream>>>(Aq, Qp, Qn, maxpos, maxneg, meta, out, 1);
  }
}
